// CapsuleLayer_5411658793439
// MI455X (gfx1250) — matrix operand order repaired and checked by reading, not compiled, not run on hardware
//
#include <hip/hip_runtime.h>

// CapsuleLayer: u_hat GEMM (bf16 WMMA, f32 accum) fused with 3-iter dynamic
// routing, one workgroup per route r. B=64, C=32, R=1152, IN=128, OUT=16.

typedef __attribute__((ext_vector_type(16))) __bf16 v16bf;
typedef __attribute__((ext_vector_type(8)))  float  v8f;

#define B_    64
#define C_    32
#define R_    1152
#define IN_   128
#define OUT_  16
#define UST   (C_*OUT_ + 4)   // 516 floats: padded per-batch stride in LDS u_hat

union V16 { v16bf v; unsigned int u[8]; };
union V8  { v8f   v; float f[8]; };

__device__ __forceinline__ unsigned int pk_bf16(float a, float b) {
  union { __bf16 h; unsigned short s; } x, y;
  x.h = (__bf16)a; y.h = (__bf16)b;
  return (unsigned int)x.s | ((unsigned int)y.s << 16);
}

__global__ __launch_bounds__(256)
void caps_route_kernel(const float* __restrict__ x,
                       const float* __restrict__ w,
                       float* __restrict__ out)
{
  extern __shared__ char smem[];
  __bf16* xsh = (__bf16*)smem;                 // [64][128] bf16 = 16 KB
  float*  ush = (float*)(smem + B_*IN_*2);     // [64][516] f32  = 129 KB

  const int tid  = threadIdx.x;
  const int r    = blockIdx.x;
  const int lane = tid & 31;
  const int wv   = tid >> 5;        // wave id 0..7
  const int half = lane >> 4;       // 0/1 half-wave
  const int l16  = lane & 15;

  // ---------------- stage x (64x128 f32) into LDS as bf16 ----------------
  #pragma unroll
  for (int i = 0; i < (B_*IN_)/256; ++i) {
    int idx = tid + i*256;
    xsh[idx] = (__bf16)x[idx];
  }
  __syncthreads();

  // ---------------- GEMM phase: u_hat[b, c, :] for this r ----------------
  // wave wv handles capsules c = 4*wv .. 4*wv+3
  for (int ci = 0; ci < 4; ++ci) {
    int c = wv*4 + ci;
    const float* wb = w + ((long)(c*R_ + r)) * (IN_*OUT_);  // 128x16 block, row-major

    // B fragments (32x16 bf16, K-chunks of 32).
    // Lane layout: col N = l16; lanes 0-15 cover K=k0..k0+15, lanes 16-31
    // cover K=k0+16..k0+31; VGPR v holds K = k0 + 16*half + 2v, +1 packed.
    V16 Bf[4];
    #pragma unroll
    for (int kc = 0; kc < 4; ++kc) {
      #pragma unroll
      for (int v = 0; v < 8; ++v) {
        int k = kc*32 + half*16 + 2*v;
        Bf[kc].u[v] = pk_bf16(wb[k*OUT_ + l16], wb[(k+1)*OUT_ + l16]);
      }
    }

    #pragma unroll
    for (int mt = 0; mt < 4; ++mt) {
      int m0 = mt*16;
      // A fragments (16x32 bf16) from LDS, same contiguous k order as B:
      // lanes 0-15: VGPR0-7 K=0..15; lanes 16-31: K=16..31.
      V16 Af[4];
      int row = m0 + l16;
      #pragma unroll
      for (int kc = 0; kc < 4; ++kc) {
        #pragma unroll
        for (int v = 0; v < 8; ++v) {
          int k = kc*32 + 2*v + 16*half;
          Af[kc].u[v] = *(const unsigned int*)&xsh[row*IN_ + k];
        }
      }
      V8 acc;
      #pragma unroll
      for (int j = 0; j < 8; ++j) acc.f[j] = 0.0f;
      #pragma unroll
      for (int kc = 0; kc < 4; ++kc)
        acc.v = __builtin_amdgcn_wmma_f32_16x16x32_bf16(
                    false, Af[kc].v, false, Bf[kc].v,
                    (short)0, acc.v, false, false);
      // C/D layout: VGPR j -> M = m0 + j + 8*half, N = l16
      #pragma unroll
      for (int j = 0; j < 8; ++j) {
        int m = m0 + j + 8*half;
        ush[m*UST + c*OUT_ + l16] = acc.f[j];
      }
    }
  }
  __syncthreads();

  // ---------------- dynamic routing (3 iters), all in LDS/regs -----------
  // 4 threads per batch b: thread quad {b, sub}, sub owns outputs 4*sub..4*sub+3
  const int b   = tid >> 2;
  const int sub = tid & 3;
  const float* ubase = ush + b*UST + sub*4;

  float logit[C_];
  #pragma unroll
  for (int c = 0; c < C_; ++c) logit[c] = 0.0f;

  float v0 = 0.f, v1 = 0.f, v2 = 0.f, v3 = 0.f;

  #pragma unroll
  for (int it = 0; it < 3; ++it) {
    // softmax over capsules c
    float mx = logit[0];
    #pragma unroll
    for (int c = 1; c < C_; ++c) mx = fmaxf(mx, logit[c]);
    float coef[C_]; float sum = 0.f;
    #pragma unroll
    for (int c = 0; c < C_; ++c) { coef[c] = __expf(logit[c] - mx); sum += coef[c]; }
    float inv = 1.0f / sum;

    // s = sum_c coef[c] * u_hat[b,c,:]  (this thread's 4 outputs)
    float s0 = 0.f, s1 = 0.f, s2 = 0.f, s3 = 0.f;
    #pragma unroll
    for (int c = 0; c < C_; ++c) {
      float4 u = *(const float4*)(ubase + c*OUT_);
      float cc = coef[c] * inv;
      s0 = fmaf(cc, u.x, s0); s1 = fmaf(cc, u.y, s1);
      s2 = fmaf(cc, u.z, s2); s3 = fmaf(cc, u.w, s3);
    }
    // squash: v = s * ||s|| / (1 + ||s||^2); norm reduced across the quad
    float p = s0*s0 + s1*s1 + s2*s2 + s3*s3;
    p += __shfl_xor(p, 1, 32);
    p += __shfl_xor(p, 2, 32);
    float sc = sqrtf(p) / (1.0f + p);
    v0 = s0*sc; v1 = s1*sc; v2 = s2*sc; v3 = s3*sc;

    if (it < 2) {
      // agreement: logit[c] += <u_hat[b,c,:], v>
      #pragma unroll
      for (int c = 0; c < C_; ++c) {
        float4 u = *(const float4*)(ubase + c*OUT_);
        float d = u.x*v0 + u.y*v1 + u.z*v2 + u.w*v3;
        d += __shfl_xor(d, 1, 32);
        d += __shfl_xor(d, 2, 32);
        logit[c] += d;
      }
    }
  }

  // out: (B, R, OUT)
  *(float4*)(out + ((long)b*R_ + r)*OUT_ + sub*4) = make_float4(v0, v1, v2, v3);
}

extern "C" void kernel_launch(void* const* d_in, const int* in_sizes, int n_in,
                              void* d_out, int out_size, void* d_ws, size_t ws_size,
                              hipStream_t stream) {
  (void)in_sizes; (void)n_in; (void)d_ws; (void)ws_size; (void)out_size;
  const float* x = (const float*)d_in[0];        // (64,1,1,128) f32
  const float* w = (const float*)d_in[1];        // (32,1152,128,16) f32
  float* out = (float*)d_out;                    // (64,1152,16) f32

  size_t shmem = (size_t)B_*IN_*2 + (size_t)B_*UST*4;   // 16384 + 132096 = 148480 B
  caps_route_kernel<<<dim3(R_), dim3(256), shmem, stream>>>(x, w, out);
}
